// MPNNGNN_87033217286678
// MI455X (gfx1250) — matrix operand order repaired and checked by reading, not compiled, not run on hardware
//
#include <hip/hip_runtime.h>

#define NN   40000
#define EE   160000
#define DD   32
#define EHD  128
#define TILES 10000          // EE / 16
#define EPSV 1e-5f

typedef __attribute__((ext_vector_type(16))) _Float16 v16h;
typedef __attribute__((ext_vector_type(8)))  float    v8f;

// Intra-wave LDS producer->consumer sync: stop compiler reordering and drain DS.
__device__ __forceinline__ void wave_sync() {
  __builtin_amdgcn_wave_barrier();
  asm volatile("s_wait_dscnt 0" ::: "memory");
  __builtin_amdgcn_wave_barrier();
}

__device__ __forceinline__ v8f wmma_f16(v16h a, v16h b, v8f c) {
  return __builtin_amdgcn_wmma_f32_16x16x32_f16(false, a, false, b, (short)0, c, false, false);
}

// ---------------------------------------------------------------------------
// Pack eW2 (f32 [128,1024]) -> f16 B-tiles in WMMA lane layout.
// Tile (i,c,half): B[kk, o'] = eW2[c*32+kk, i*32+half*16+o'],
// lane L holds K = (L>=16?8:0)+(t<8?t:t+8), N = L&15.  Linear: tid = tileidx*512+L*16+t.
__global__ __launch_bounds__(256) void k_pack_w2(const float* __restrict__ eW2,
                                                 const float* __restrict__ eb2,
                                                 _Float16* __restrict__ pw2,
                                                 _Float16* __restrict__ peb2) {
  int tid = blockIdx.x * 256 + threadIdx.x;
  if (tid < 131072) {
    int t = tid & 15, L = (tid >> 4) & 31, tilei = tid >> 9;
    int half = tilei & 1, c = (tilei >> 1) & 3, i = tilei >> 3;
    int K   = c * 32 + ((L >= 16) ? 8 : 0) + (t < 8 ? t : t + 8);
    int col = i * 32 + half * 16 + (L & 15);
    pw2[tid] = (_Float16)eW2[K * 1024 + col];
  }
  if (tid < 1024) {   // eb2 as two B tiles (K=i dim)
    int t = tid & 15, L = (tid >> 4) & 31, half = (tid >> 9) & 1;
    int K = ((L >= 16) ? 8 : 0) + (t < 8 ? t : t + 8);
    peb2[tid] = (_Float16)eb2[K * 32 + half * 16 + (L & 15)];
  }
}

// Fold BatchNorm2 into eW1/eb1.
__global__ __launch_bounds__(256) void k_pack_w1(const float* __restrict__ eW1,
                                                 const float* __restrict__ eb1,
                                                 const float* __restrict__ g2,
                                                 const float* __restrict__ b2,
                                                 float* __restrict__ pW1,
                                                 float* __restrict__ pb1) {
  int tid = blockIdx.x * 256 + threadIdx.x;
  float inv = rsqrtf(1.f + EPSV);
  if (tid < 2048) {
    int k = tid & 127;
    pW1[tid] = eW1[tid] * g2[k] * inv;
  }
  if (tid < 128) pb1[tid] = eb1[tid] * g2[tid] * inv + b2[tid];
}

// Node projection: state = relu(BN(x @ projW + b)). One lane per (node, channel).
__global__ __launch_bounds__(256) void k_proj(const float* __restrict__ x,
                                              const float* __restrict__ W,
                                              const float* __restrict__ b,
                                              const float* __restrict__ g,
                                              const float* __restrict__ bb,
                                              float* __restrict__ state) {
  int tid = blockIdx.x * 256 + threadIdx.x;
  int n = tid >> 5, o = tid & 31;
  const float* xr = x + n * 64;
  float acc = 0.f;
  for (int j = 0; j < 64; ++j) acc = fmaf(xr[j], W[j * 32 + o], acc);
  float v = (acc + b[o]) * (g[o] * rsqrtf(1.f + EPSV)) + bb[o];
  state[tid] = v > 0.f ? v : 0.f;
}

// eh = relu(BN(edge_attr @ eW1)), stored f16 in WMMA A-lane layout:
// ehp[tile*2048 + c*512 + L*16 + t],  k = c*32 + (L>=16?8:0) + (t<8? t : t+8)
__global__ __launch_bounds__(256) void k_eh(const float* __restrict__ ea,
                                            const float* __restrict__ pW1,
                                            const float* __restrict__ pb1,
                                            _Float16* __restrict__ ehp) {
  int tid = blockIdx.x * 256 + threadIdx.x;   // TILES*128 threads exactly
  int L = tid & 31, c = (tid >> 5) & 3, tile = tid >> 7;
  int e = tile * 16 + (L & 15);
  float ar[16];
  const float* row = ea + e * 16;
  for (int j = 0; j < 16; ++j) ar[j] = row[j];
  int off = (L >= 16) ? 8 : 0;
  v16h out;
  for (int t = 0; t < 16; ++t) {
    int k = c * 32 + off + (t < 8 ? t : t + 8);
    float acc = pb1[k];
    for (int j = 0; j < 16; ++j) acc = fmaf(ar[j], pW1[j * 128 + k], acc);
    out[t] = (_Float16)(acc > 0.f ? acc : 0.f);
  }
  *(v16h*)(ehp + (size_t)tid * 16) = out;
}

__global__ __launch_bounds__(256) void k_zero(float* __restrict__ p, int n) {
  int t = blockIdx.x * 256 + threadIdx.x;
  if (t < n) p[t] = 0.f;
}

// ---------------------------------------------------------------------------
// Fused edge-message kernel: one wave = 16 edges.
// msg = P @ W2'  with  P[e,(k,i)] = eh[e,k]*h_src[e,i]  built as outer product,
// plus an eb2 K-chunk (A = h tile).  258 v_wmma per tile, eW2' resident in LDS.
__global__ __launch_bounds__(256) void k_msg(const float* __restrict__ state,
                                             const int* __restrict__ src,
                                             const int* __restrict__ dst,
                                             const _Float16* __restrict__ pw2,
                                             const _Float16* __restrict__ peb2,
                                             const _Float16* __restrict__ ehp,
                                             float* __restrict__ agg) {
  extern __shared__ __align__(32) unsigned char smemraw[];
  _Float16* sW2  = (_Float16*)smemraw;       // 131072 halfs = 256 KB
  _Float16* sEb2 = sW2 + 131072;             // 1024 halfs
  _Float16* sH   = sEb2 + 1024;              // 8 waves * 544 halfs (stride 34: bank-safe)
  int*      sDst = (int*)(sH + 8 * 544);     // 8 * 16

  int tid = threadIdx.x;
  { // cooperative LDS fill of packed weights
    const uint4* s4 = (const uint4*)pw2;  uint4* d4 = (uint4*)sW2;
    for (int i = tid; i < 131072 / 8; i += 256) d4[i] = s4[i];
    const uint4* e4 = (const uint4*)peb2; uint4* f4 = (uint4*)sEb2;
    for (int i = tid; i < 1024 / 8; i += 256) f4[i] = e4[i];
  }
  __syncthreads();

  int w = tid >> 5, lane = tid & 31;
  int eloc = lane & 15, halfId = lane >> 4;
  _Float16* myH = sH + w * 544;
  int* myDst = sDst + w * 16;

  for (int tile = blockIdx.x * 8 + w; tile < TILES; tile += gridDim.x * 8) {
    int eg = tile * 16 + eloc;
    { // gather h_src tile -> LDS (f16), lane covers half a row
      int s = src[eg];
      const float4* hp = (const float4*)(state + s * 32 + halfId * 16);
      for (int q = 0; q < 4; ++q) {
        float4 v = hp[q];
        int base = eloc * 34 + halfId * 16 + q * 4;
        myH[base + 0] = (_Float16)v.x; myH[base + 1] = (_Float16)v.y;
        myH[base + 2] = (_Float16)v.z; myH[base + 3] = (_Float16)v.w;
      }
      if (halfId == 0) myDst[eloc] = dst[eg];
    }
    // eh chunks, already in A-lane layout
    const _Float16* ehb = ehp + (size_t)tile * 2048 + lane * 16;
    v16h eh0 = *(const v16h*)(ehb);
    v16h eh1 = *(const v16h*)(ehb + 512);
    v16h eh2 = *(const v16h*)(ehb + 1024);
    v16h eh3 = *(const v16h*)(ehb + 1536);
    wave_sync();

    // A-layout h tile for the eb2 chunk
    v16h hA;
    for (int t = 0; t < 16; ++t) {
      int i = halfId * 8 + (t < 8 ? t : t + 8);
      hA[t] = myH[eloc * 34 + i];
    }
    v8f acc0 = {0,0,0,0,0,0,0,0};
    v8f acc1 = {0,0,0,0,0,0,0,0};
    acc0 = wmma_f16(hA, *(const v16h*)(sEb2 + lane * 16),       acc0);
    acc1 = wmma_f16(hA, *(const v16h*)(sEb2 + 512 + lane * 16), acc1);

    for (int i = 0; i < 32; ++i) {
      _Float16 hs = myH[eloc * 34 + i];      // per-edge scalar h_src[e,i]
      const _Float16* bp = sW2 + (size_t)i * 4096 + lane * 16;
      v16h A;
      A = eh0 * hs;
      acc0 = wmma_f16(A, *(const v16h*)(bp +    0), acc0);
      acc1 = wmma_f16(A, *(const v16h*)(bp +  512), acc1);
      A = eh1 * hs;
      acc0 = wmma_f16(A, *(const v16h*)(bp + 1024), acc0);
      acc1 = wmma_f16(A, *(const v16h*)(bp + 1536), acc1);
      A = eh2 * hs;
      acc0 = wmma_f16(A, *(const v16h*)(bp + 2048), acc0);
      acc1 = wmma_f16(A, *(const v16h*)(bp + 2560), acc1);
      A = eh3 * hs;
      acc0 = wmma_f16(A, *(const v16h*)(bp + 3072), acc0);
      acc1 = wmma_f16(A, *(const v16h*)(bp + 3584), acc1);
    }
    // scatter-add using C/D layout: VGPR r -> edge row r (+8 for upper lanes)
    for (int r = 0; r < 8; ++r) {
      int d = myDst[r + halfId * 8];
      unsafeAtomicAdd(agg + d * 32 + eloc,      acc0[r]);
      unsafeAtomicAdd(agg + d * 32 + 16 + eloc, acc1[r]);
    }
    wave_sync();
  }
}

// ---------------------------------------------------------------------------
// Node update: conv = agg + state@root + cb ; relu ; GRU cell ; agg reset.
__global__ __launch_bounds__(256) void k_update(float* __restrict__ state,
                                                float* __restrict__ agg,
                                                const float* __restrict__ root,
                                                const float* __restrict__ cb,
                                                const float* __restrict__ Wih,
                                                const float* __restrict__ Whh,
                                                const float* __restrict__ bih,
                                                const float* __restrict__ bhh,
                                                float* __restrict__ outp) {
  __shared__ float sRoot[1024];
  __shared__ float sWih[3072];
  __shared__ float sWhh[3072];
  __shared__ float sm[8][32];
  __shared__ float shd[8][32];
  int tid = threadIdx.x;
  for (int i = tid; i < 1024; i += 256) sRoot[i] = root[i];
  for (int i = tid; i < 3072; i += 256) { sWih[i] = Wih[i]; sWhh[i] = Whh[i]; }

  int ty = tid >> 5, o = tid & 31;
  int n = blockIdx.x * 8 + ty;
  float hid = state[n * 32 + o];
  shd[ty][o] = hid;
  float acc = cb[o] + agg[n * 32 + o];
  __syncthreads();
  for (int i = 0; i < 32; ++i) acc = fmaf(shd[ty][i], sRoot[i * 32 + o], acc);
  float m = acc > 0.f ? acc : 0.f;
  sm[ty][o] = m;
  __syncthreads();
  float gir = bih[o], giz = bih[32 + o], gin = bih[64 + o];
  float ghr = bhh[o], ghz = bhh[32 + o], ghn = bhh[64 + o];
  for (int i = 0; i < 32; ++i) {
    float mi = sm[ty][i], hi = shd[ty][i];
    gir = fmaf(mi, sWih[o * 32 + i],        gir);
    giz = fmaf(mi, sWih[(32 + o) * 32 + i], giz);
    gin = fmaf(mi, sWih[(64 + o) * 32 + i], gin);
    ghr = fmaf(hi, sWhh[o * 32 + i],        ghr);
    ghz = fmaf(hi, sWhh[(32 + o) * 32 + i], ghz);
    ghn = fmaf(hi, sWhh[(64 + o) * 32 + i], ghn);
  }
  float r = 1.f / (1.f + __expf(-(gir + ghr)));
  float z = 1.f / (1.f + __expf(-(giz + ghz)));
  float nn = tanhf(gin + r * ghn);
  float hnew = (1.f - z) * nn + z * hid;
  state[n * 32 + o] = hnew;
  agg[n * 32 + o] = 0.f;     // reset for next step's scatter
  outp[n * 32 + o] = hnew;   // last step's write is the final output
}

// ---------------------------------------------------------------------------
extern "C" void kernel_launch(void* const* d_in, const int* in_sizes, int n_in,
                              void* d_out, int out_size, void* d_ws, size_t ws_size,
                              hipStream_t stream) {
  const float* x     = (const float*)d_in[0];
  const int*   eidx  = (const int*)  d_in[1];
  const float* ea    = (const float*)d_in[2];
  const float* projW = (const float*)d_in[3];
  const float* projb = (const float*)d_in[4];
  const float* bn1g  = (const float*)d_in[5];
  const float* bn1b  = (const float*)d_in[6];
  const float* eW1   = (const float*)d_in[7];
  const float* eb1   = (const float*)d_in[8];
  const float* bn2g  = (const float*)d_in[9];
  const float* bn2b  = (const float*)d_in[10];
  const float* eW2   = (const float*)d_in[11];
  const float* eb2   = (const float*)d_in[12];
  const float* root  = (const float*)d_in[13];
  const float* cb    = (const float*)d_in[14];
  const float* Wih   = (const float*)d_in[15];
  const float* Whh   = (const float*)d_in[16];
  const float* bih   = (const float*)d_in[17];
  const float* bhh   = (const float*)d_in[18];

  unsigned char* ws = (unsigned char*)d_ws;
  float*    state = (float*)(ws);                    // 5,120,000 B
  float*    agg   = (float*)(ws + 5120000);          // 5,120,000 B
  _Float16* pw2   = (_Float16*)(ws + 10240000);      //   262,144 B
  _Float16* peb2  = (_Float16*)(ws + 10502144);      //     2,048 B
  float*    pW1   = (float*)(ws + 10504192);         //     8,192 B
  float*    pb1   = (float*)(ws + 10512384);         //       512 B
  _Float16* ehp   = (_Float16*)(ws + 10512896);      // 40,960,000 B  (~51.5 MB total)

  const int* src  = eidx;
  const int* dstp = eidx + EE;

  k_pack_w2<<<512, 256, 0, stream>>>(eW2, eb2, pw2, peb2);
  k_pack_w1<<<9, 256, 0, stream>>>(eW1, eb1, bn2g, bn2b, pW1, pb1);
  k_proj<<<5000, 256, 0, stream>>>(x, projW, projb, bn1g, bn1b, state);
  k_zero<<<5000, 256, 0, stream>>>(agg, NN * 32);
  k_eh<<<5000, 256, 0, stream>>>(ea, pW1, pb1, ehp);

  size_t ldsBytes = 131072 * 2 + 1024 * 2 + 8 * 544 * 2 + 8 * 16 * 4;  // 273,408 B
  for (int s = 0; s < 3; ++s) {
    k_msg<<<128, 256, ldsBytes, stream>>>(state, src, dstp, pw2, peb2, ehp, agg);
    k_update<<<5000, 256, 0, stream>>>(state, agg, root, cb, Wih, Whh, bih, bhh,
                                       (float*)d_out);
  }
}
